// SwinSelfAttention_15212774163131
// MI455X (gfx1250) — matrix operand order repaired and checked by reading, not compiled, not run on hardware
//
#include <hip/hip_runtime.h>
#include <hip/hip_bf16.h>

typedef __attribute__((ext_vector_type(16))) __bf16 v16bf;
typedef __attribute__((ext_vector_type(8)))  __bf16 v8bf;
typedef __attribute__((ext_vector_type(4)))  __bf16 v4bf;
typedef __attribute__((ext_vector_type(8)))  float  v8f;
typedef __attribute__((ext_vector_type(4)))  float  v4f;

#define WMMA_BF16(a, b, c) \
    __builtin_amdgcn_wmma_f32_16x16x32_bf16(false, (a), false, (b), (short)0, (c), false, false)

__device__ __forceinline__ v16bf combine16(v8bf lo, v8bf hi) {
    v16bf r;
#pragma unroll
    for (int i = 0; i < 8; ++i) { r[i] = lo[i]; r[i + 8] = hi[i]; }
    return r;
}

// A-operand (16x32 bf16): this lane needs 16 contiguous K values.
__device__ __forceinline__ v16bf ldsA(const __bf16* p) {
    return combine16(*(const v8bf*)p, *(const v8bf*)(p + 8));
}
// B-operand (32x16 bf16): this lane needs 16 contiguous K values.
__device__ __forceinline__ v16bf ldsB(const __bf16* p) {
    return combine16(*(const v8bf*)p, *(const v8bf*)(p + 8));
}

// ---- Prep kernel: one-time f32 -> bf16 conversion of qkv_w into workspace ----
__global__ __launch_bounds__(256) void cvt_weights_kernel(const float* __restrict__ w,
                                                          __bf16* __restrict__ o, int n) {
    int i = blockIdx.x * 256 + threadIdx.x;
    if (i < n) o[i] = (__bf16)w[i];
}

// One workgroup = 64 threads = 2 waves; wave w handles head (blockIdx.y*2 + w)
// of window blockIdx.x.  Fully fused: QKV proj -> scores(+bias+mask) -> softmax -> attn@V.
__global__ __launch_bounds__(64) void swin_window_attn_kernel(
    const float*  __restrict__ hidden,      // [B, 49, 256]
    const float*  __restrict__ amask,       // [64, 49, 49]
    const __bf16* __restrict__ wbf,         // [768, 256] bf16 (pre-converted, [N][K] layout)
    const float*  __restrict__ qkv_b,       // [768]
    const float*  __restrict__ bias_table,  // [169, 8]
    const int*    __restrict__ rel_index,   // [49, 49]
    float*        __restrict__ out)         // [B, 49, 256]
{
    constexpr int N = 49, C = 256, HD = 32;
    const float SCALE = 0.17677669529663687f;  // 32^-0.5

    const int b     = blockIdx.x;
    const int w     = threadIdx.x >> 5;
    const int lane  = threadIdx.x & 31;
    const int head  = blockIdx.y * 2 + w;
    const int lhalf = lane >> 4;   // 0 or 1: which K-half this lane supplies
    const int l16   = lane & 15;

    // 56 KB static LDS with phase-based reuse:
    //  [    0,32768): phase0/1: hidden bf16 [64][256]; phase2/3: scores f32 [2][64][64]
    //  [32768,40960): qLds bf16 [2][64][32]   }  phase3/4: attn bf16 [2][64][64]
    //  [40960,49152): kLds bf16 [2][64][32]   }
    //  [49152,57344): vT  bf16 [2][32][64]
    __shared__ __align__(16) char smem[57344];
    __bf16* hidLds = (__bf16*)smem;                              // [64][256]
    float*  scLds  = (float*)(smem) + w * 4096;                  // [64][64]
    __bf16* qLds   = (__bf16*)(smem + 32768) + w * 2048;         // [64][32]
    __bf16* kLds   = (__bf16*)(smem + 40960) + w * 2048;         // [64][32]
    __bf16* vTLds  = (__bf16*)(smem + 49152) + w * 2048;         // [32][64]
    __bf16* atLds  = (__bf16*)(smem + 32768) + w * 4096;         // [64][64]

    // ---- Phase 0: stage hidden window as bf16 (float4 granularity), zero rows 49..63 ----
    for (int chunk = threadIdx.x; chunk < 64 * C / 4; chunk += 64) {
        int r  = chunk >> 6;          // 64 float4-chunks per row
        int c4 = (chunk & 63) * 4;
        v4f x = {};
        if (r < N) x = *(const v4f*)(hidden + ((size_t)b * N + r) * C + c4);
        v4bf o;
#pragma unroll
        for (int k = 0; k < 4; ++k) o[k] = (__bf16)x[k];
        *(v4bf*)(hidLds + r * C + c4) = o;
    }
    __syncthreads();

    // ---- Phase 1: QKV projection for this head (q,k -> LDS; v -> transposed LDS) ----
#pragma unroll
    for (int part = 0; part < 3; ++part) {           // 0=q, 1=k, 2=v
        const int gcol0 = part * C + head * HD;
        for (int nt = 0; nt < 2; ++nt) {
            const int   wcol = gcol0 + nt * 16 + l16;     // this lane's output column
            const float bias = qkv_b[wcol];
            // Hoist all 8 k-step B tiles (bf16 weights, 16 contiguous K per lane).
            v16bf bm[8];
#pragma unroll
            for (int kk = 0; kk < 8; ++kk)
                bm[kk] = *(const v16bf*)(wbf + (size_t)wcol * C + kk * 32 + lhalf * 16);
            for (int mt = 0; mt < 4; ++mt) {
                v8f acc = {};
#pragma unroll
                for (int kk = 0; kk < 8; ++kk) {          // K = 256 in steps of 32
                    v16bf a = ldsA(hidLds + (mt * 16 + l16) * C + kk * 32 + lhalf * 16);
                    acc = WMMA_BF16(a, bm[kk], acc);
                }
                const int cn = nt * 16 + l16;             // 0..31 within this head slice
#pragma unroll
                for (int r = 0; r < 8; ++r) {
                    int row = mt * 16 + r + lhalf * 8;
                    __bf16 val = (__bf16)(acc[r] + bias);
                    if (part == 0)      qLds[row * HD + cn] = val;
                    else if (part == 1) kLds[row * HD + cn] = val;
                    else                vTLds[cn * 64 + row] = val;   // store V transposed
                }
            }
        }
    }
    __syncthreads();   // before scores overwrite the hidden region

    // ---- Phase 2: scores = (q kT)*scale + rel_pos_bias + window mask -> f32 LDS ----
    const float* mrow = amask + (size_t)(b & 63) * N * N;
    for (int mt = 0; mt < 4; ++mt) {
        for (int nt = 0; nt < 4; ++nt) {
            v8f acc = {};
            v16bf a  = ldsA(qLds + (mt * 16 + l16) * HD + lhalf * 16);
            v16bf bm = ldsB(kLds + (nt * 16 + l16) * HD + lhalf * 16);
            acc = WMMA_BF16(a, bm, acc);
            const int j = nt * 16 + l16;
#pragma unroll
            for (int r = 0; r < 8; ++r) {
                int i = mt * 16 + r + lhalf * 8;
                float s = -1e30f;                         // pad cols vanish in softmax
                if (i < N && j < N) {
                    int ri = rel_index[i * N + j];
                    s = acc[r] * SCALE + bias_table[ri * 8 + head] + mrow[i * N + j];
                }
                scLds[i * 64 + j] = s;
            }
        }
    }
    __syncthreads();   // attn overwrites q/k regions; make sure both waves left phase 2

    // ---- Phase 3: softmax rows in registers (each lane: 2 rows), write bf16 attn ----
    for (int rr = 0; rr < 2; ++rr) {
        int row = lane + rr * 32;
        float rbuf[64];
#pragma unroll
        for (int jj = 0; jj < 16; ++jj)
            *(v4f*)&rbuf[jj * 4] = *(const v4f*)(scLds + row * 64 + jj * 4);
        float mx = rbuf[0];
#pragma unroll
        for (int j = 1; j < 64; ++j) mx = fmaxf(mx, rbuf[j]);
        float sum = 0.0f;
#pragma unroll
        for (int j = 0; j < 64; ++j) { rbuf[j] = __expf(rbuf[j] - mx); sum += rbuf[j]; }
        float inv = 1.0f / sum;
#pragma unroll
        for (int jj = 0; jj < 16; ++jj) {
            v4bf o;
#pragma unroll
            for (int k = 0; k < 4; ++k) o[k] = (__bf16)(rbuf[jj * 4 + k] * inv);
            *(v4bf*)(atLds + row * 64 + jj * 4) = o;
        }
    }
    // phase 3 -> 4 is wave-private (same-wave LDS ops are in-order); no barrier needed

    // ---- Phase 4: ctx = attn @ v ; write output rows < 49 ----
    for (int mt = 0; mt < 4; ++mt) {
        for (int nt = 0; nt < 2; ++nt) {
            v8f acc = {};
#pragma unroll
            for (int kk = 0; kk < 2; ++kk) {              // K = 64 (pad cols are 0)
                v16bf a  = ldsA(atLds + (mt * 16 + l16) * 64 + kk * 32 + lhalf * 16);
                v16bf bm = ldsB(vTLds + (nt * 16 + l16) * 64 + kk * 32 + lhalf * 16);
                acc = WMMA_BF16(a, bm, acc);
            }
            const int cn = nt * 16 + l16;
#pragma unroll
            for (int r = 0; r < 8; ++r) {
                int i = mt * 16 + r + lhalf * 8;
                if (i < N)
                    out[((size_t)b * N + i) * C + head * HD + cn] = acc[r];
            }
        }
    }
}

extern "C" void kernel_launch(void* const* d_in, const int* in_sizes, int n_in,
                              void* d_out, int out_size, void* d_ws, size_t ws_size,
                              hipStream_t stream) {
    const float* hidden     = (const float*)d_in[0];
    const float* amask      = (const float*)d_in[1];
    const float* qkv_w      = (const float*)d_in[2];
    const float* qkv_b      = (const float*)d_in[3];
    const float* bias_table = (const float*)d_in[4];
    const int*   rel_index  = (const int*)d_in[5];
    float*       out        = (float*)d_out;

    // Workspace: bf16 copy of qkv_w (768*256*2 = 393216 bytes).
    __bf16* wbf = (__bf16*)d_ws;
    const int nw = in_sizes[2];                  // 196608 weight elements
    cvt_weights_kernel<<<(nw + 255) / 256, 256, 0, stream>>>(qkv_w, wbf, nw);

    const int Bwin = in_sizes[0] / (49 * 256);   // 4096 windows
    dim3 grid(Bwin, 4);                          // y: 4 blocks x 2 waves = 8 heads
    dim3 block(64);                              // 2 waves of 32
    swin_window_attn_kernel<<<grid, block, 0, stream>>>(
        hidden, amask, wbf, qkv_b, bias_table, rel_index, out);
}
